// SlotGATTest2_90031104459544
// MI455X (gfx1250) — matrix operand order repaired and checked by reading, not compiled, not run on hardware
//
#include <hip/hip_runtime.h>
#include <hip/hip_bf16.h>
#include <cstdint>
#include <cstddef>

typedef __attribute__((ext_vector_type(16))) _Float16 v16h;
typedef __attribute__((ext_vector_type(8)))  float    v8f;

// Problem constants (fixed by the reference)
static constexpr int C_  = 8;
static constexpr int N_  = 1024;
static constexpr int T_  = 8;
static constexpr int I_  = 64;
static constexpr int H_  = 4;
static constexpr int F_  = 64;
static constexpr int TF_ = T_ * F_;       // 512
static constexpr int CN_ = C_ * N_;       // 8192
static constexpr float NEG_SLOPE = 0.2f;

// Padded column stride for the transposed B tile in LDS:
// 72 halves = 144 B = 36 words -> 16 consecutive cols map to 16 distinct banks.
static constexpr int BSTRIDE = 72;

__device__ __forceinline__ float lrelu(float x) { return x >= 0.f ? x : NEG_SLOPE * x; }

// ---------------------------------------------------------------------------
// K1: build adjacency bitmasks (row-major adj[i][jbits] and transposed adjT[j][ibits])
// ---------------------------------------------------------------------------
__global__ __launch_bounds__(256) void build_adj_kernel(const int* __restrict__ edges,
                                                        unsigned* __restrict__ adj,
                                                        unsigned* __restrict__ adjT,
                                                        int E) {
  int e = blockIdx.x * 256 + threadIdx.x;
  if (e >= E) return;
  int s = edges[e];
  int d = edges[E + e];
  int c = s >> 10;          // src conformer (dst is in the same conformer by construction)
  int i = s & 1023;
  int j = d & 1023;
  atomicOr(&adj [((size_t)(c * N_ + i)) * 32 + (j >> 5)], 1u << (j & 31));
  atomicOr(&adjT[((size_t)(c * N_ + j)) * 32 + (i >> 5)], 1u << (i & 31));
}

// ---------------------------------------------------------------------------
// K2: per-type GEMM feat0 = node_feats[c,:,t,:] @ fc[t]  (f16 in, f32 acc, WMMA)
//     with the torch reshape/permute scramble fused into the scatter-store:
//     output (row n, col k=h*64+f) of type t, conformer c lands at
//       feat[((c*1024 + t*128 + n/8)*4 + h)*512 + (n%8)*64 + f]
// ---------------------------------------------------------------------------
__device__ __forceinline__ v16h load_a_frag(const _Float16* __restrict__ sA,
                                            int rowBase, int lane, int kc) {
  // 16-bit A 16x32 layout (ISA 7.12.2): lane m=lane%16; elements e=0..7 -> K=kb+e,
  // e=8..15 -> K=16+kb+(e-8), kb = (lane<16)?0:8. kc selects K chunk (0..31 / 32..63).
  int m  = lane & 15;
  int kb = (lane < 16) ? 0 : 8;
  const _Float16* rowp = sA + (rowBase + m) * 64 + kc * 32;
  union { uint4 u[2]; v16h v; } r;
  // two 16B-aligned contiguous 8-half segments -> ds_load_b128 x2
  r.u[0] = *reinterpret_cast<const uint4*>(rowp + kb);        // K = kb..kb+7
  r.u[1] = *reinterpret_cast<const uint4*>(rowp + 16 + kb);   // K = 16+kb..16+kb+7
  return r.v;
}

__device__ __forceinline__ v16h load_b_frag(const _Float16* __restrict__ sBt,
                                            int colBase, int lane, int kc) {
  // 16-bit B 32x16 layout: col = lane%16; element e=0..7 -> K = kb + e, e=8..15 -> K = kb+16+(e-8),
  // kb = kc*32 + ((lane<16)?0:8). sBt is col-major [col][BSTRIDE] so the
  // 16 K-halves are contiguous and 16B-aligned -> ds_load_b128 x2.
  int col = colBase + (lane & 15);
  int kb  = kc * 32 + ((lane < 16) ? 0 : 8);
  const _Float16* p = sBt + col * BSTRIDE + kb;
  union { uint4 u[2]; v16h v; } r;
  r.u[0] = *reinterpret_cast<const uint4*>(p);
  r.u[1] = *reinterpret_cast<const uint4*>(p + 16);
  return r.v;
}

__global__ __launch_bounds__(256) void gemm_feat_kernel(const float* __restrict__ nf,
                                                        const float* __restrict__ fc,
                                                        float* __restrict__ feat) {
  // blockIdx.x = (c*8 + t)*16 + mb ; each block does 64 rows x 256 cols, K=64
  int bid = blockIdx.x;
  int mb = bid & 15;
  int t  = (bid >> 4) & 7;
  int c  = bid >> 7;
  int n0 = mb * 64;

  __shared__ _Float16 sA[64 * 64];            //  8 KB, row-major [row][64]
  __shared__ _Float16 sBt[256 * BSTRIDE];     // 36 KB, col-major [col][BSTRIDE]

  int tid = threadIdx.x;

  // Stage A tile (rows n0..n0+63 of type t), converting f32 -> f16.
  {
    int r  = tid >> 2;            // 0..63
    int c0 = (tid & 3) * 16;      // 0,16,32,48
    const float* src = nf + (((size_t)(c * N_ + n0 + r) * T_ + t) * I_) + c0;
    _Float16* dst = sA + r * 64 + c0;
#pragma unroll
    for (int e = 0; e < 16; ++e) dst[e] = (_Float16)src[e];
  }
  // Stage B tile fc[t] (64 x 256) transposed: thread tid owns output column tid.
  // Global reads are coalesced across the block (consecutive tid -> consecutive col),
  // LDS writes are 8x b128 per thread.
  {
    const float* src = fc + (size_t)t * I_ * 256 + tid;
    _Float16* dst = sBt + tid * BSTRIDE;
#pragma unroll
    for (int ib = 0; ib < 64; ib += 8) {
      _Float16 tmp[8];
#pragma unroll
      for (int e = 0; e < 8; ++e) tmp[e] = (_Float16)src[(size_t)(ib + e) * 256];
      *reinterpret_cast<uint4*>(dst + ib) = *reinterpret_cast<const uint4*>(tmp);
    }
  }
  __syncthreads();

  int wave = tid >> 5, lane = tid & 31;
  v8f acc[4][2];
#pragma unroll
  for (int rt = 0; rt < 4; ++rt)
#pragma unroll
    for (int ct = 0; ct < 2; ++ct)
#pragma unroll
      for (int r = 0; r < 8; ++r) acc[rt][ct][r] = 0.f;

#pragma unroll
  for (int kc = 0; kc < 2; ++kc) {
    v16h b0 = load_b_frag(sBt, (2 * wave + 0) * 16, lane, kc);
    v16h b1 = load_b_frag(sBt, (2 * wave + 1) * 16, lane, kc);
#pragma unroll
    for (int rt = 0; rt < 4; ++rt) {
      v16h af = load_a_frag(sA, rt * 16, lane, kc);
      acc[rt][0] = __builtin_amdgcn_wmma_f32_16x16x32_f16(false, af, false, b0,
                                                          (short)0, acc[rt][0], false, false);
      acc[rt][1] = __builtin_amdgcn_wmma_f32_16x16x32_f16(false, af, false, b1,
                                                          (short)0, acc[rt][1], false, false);
    }
  }

  // Scatter-store with the scramble. D layout: VGPR r holds M=r (lanes 0-15)
  // or M=8+r (lanes 16-31); col = lane%16.
#pragma unroll
  for (int rt = 0; rt < 4; ++rt) {
#pragma unroll
    for (int ct = 0; ct < 2; ++ct) {
      int kcol = (wave * 2 + ct) * 16 + (lane & 15);   // 0..255
      int h = kcol >> 6, f = kcol & 63;
#pragma unroll
      for (int r = 0; r < 8; ++r) {
        int mloc = r + ((lane < 16) ? 0 : 8);
        int n = n0 + rt * 16 + mloc;
        size_t addr = ((size_t)((c * N_ + t * 128 + (n >> 3)) * H_ + h)) * (size_t)TF_
                      + (size_t)(n & 7) * 64 + f;
        feat[addr] = acc[rt][ct][r];
      }
    }
  }
}

// ---------------------------------------------------------------------------
// K3: a_src/a_dst — per (node,h) dot of feat row with attention vectors.
// One wave per (c*N+n, h).
// ---------------------------------------------------------------------------
__global__ __launch_bounds__(256) void attn_vec_kernel(const float* __restrict__ feat,
                                                       const float* __restrict__ attn_s,
                                                       const float* __restrict__ attn_d,
                                                       float* __restrict__ a_src,
                                                       float* __restrict__ a_dst) {
  int gw   = blockIdx.x * 8 + (threadIdx.x >> 5);   // node*4 + h, 0..32767
  int lane = threadIdx.x & 31;
  int h    = gw & 3;
  int node = gw >> 2;                               // c*N + n
  const float* fp = feat + ((size_t)node * H_ + h) * TF_;
  const float* as = attn_s + h * TF_;
  const float* ad = attn_d + h * TF_;
  float s0 = 0.f, s1 = 0.f;
#pragma unroll
  for (int e = lane; e < TF_; e += 32) {
    float v = fp[e];
    s0 += v * as[e];
    s1 += v * ad[e];
  }
#pragma unroll
  for (int off = 16; off; off >>= 1) {
    s0 += __shfl_xor(s0, off, 32);
    s1 += __shfl_xor(s1, off, 32);
  }
  if (lane == 0) { a_src[gw] = s0; a_dst[gw] = s1; }
}

// ---------------------------------------------------------------------------
// K4: per-column softmax stats. One wave per column (c*N+j):
//   m[h]   = leakyrelu(max_{i in col} a_src[i,h] + a_dst[j,h])
//   invZ[h]= 1 / sum_{i in col} exp(leakyrelu(a_src[i,h]+a_dst[j,h]) - m[h])
// ---------------------------------------------------------------------------
__global__ __launch_bounds__(256) void colstats_kernel(const unsigned* __restrict__ adjT,
                                                       const float* __restrict__ a_src,
                                                       const float* __restrict__ a_dst,
                                                       float* __restrict__ colm,
                                                       float* __restrict__ colinv) {
  int gw   = blockIdx.x * 8 + (threadIdx.x >> 5);   // c*N + j
  int lane = threadIdx.x & 31;
  int c    = gw >> 10;
  unsigned word = adjT[(size_t)gw * 32 + lane];
  int ibase = (c << 10) + (lane << 5);              // flat source node base

  float mx[4] = {-3.0e38f, -3.0e38f, -3.0e38f, -3.0e38f};
  unsigned w = word;
  while (w) {
    int b = __ffs(w) - 1; w &= w - 1;
    const float* ap = a_src + (size_t)(ibase + b) * 4;
#pragma unroll
    for (int h = 0; h < 4; ++h) mx[h] = fmaxf(mx[h], ap[h]);
  }
#pragma unroll
  for (int off = 16; off; off >>= 1)
#pragma unroll
    for (int h = 0; h < 4; ++h) mx[h] = fmaxf(mx[h], __shfl_xor(mx[h], off, 32));

  float ad[4], mf[4];
#pragma unroll
  for (int h = 0; h < 4; ++h) ad[h] = a_dst[(size_t)gw * 4 + h];
  bool empty = (mx[0] < -1.0e37f);                  // uniform across the wave
#pragma unroll
  for (int h = 0; h < 4; ++h) mf[h] = empty ? 0.f : lrelu(mx[h] + ad[h]);

  float sm[4] = {0.f, 0.f, 0.f, 0.f};
  if (!empty) {
    w = word;
    while (w) {
      int b = __ffs(w) - 1; w &= w - 1;
      const float* ap = a_src + (size_t)(ibase + b) * 4;
#pragma unroll
      for (int h = 0; h < 4; ++h) sm[h] += __expf(lrelu(ap[h] + ad[h]) - mf[h]);
    }
  }
#pragma unroll
  for (int off = 16; off; off >>= 1)
#pragma unroll
    for (int h = 0; h < 4; ++h) sm[h] += __shfl_xor(sm[h], off, 32);

  if (lane == 0) {
#pragma unroll
    for (int h = 0; h < 4; ++h) {
      colm  [(size_t)gw * 4 + h] = mf[h];
      colinv[(size_t)gw * 4 + h] = empty ? 0.f : 1.f / sm[h];
    }
  }
}

// ---------------------------------------------------------------------------
// K5: dense attn writer — attn[c,i,j,h] = adj ? exp(lr(a_src+a_dst)-m)*invZ : 0.
// One block per source row (c*N+i); float4 (b128) coalesced stores.
// ---------------------------------------------------------------------------
__global__ __launch_bounds__(256) void attn_write_kernel(const unsigned* __restrict__ adj,
                                                         const float* __restrict__ a_src,
                                                         const float* __restrict__ a_dst,
                                                         const float* __restrict__ colm,
                                                         const float* __restrict__ colinv,
                                                         float* __restrict__ attn) {
  int row  = blockIdx.x;                 // c*N + i
  int c    = row >> 10;
  int lane = threadIdx.x & 31, wave = threadIdx.x >> 5;
  float as[4];
#pragma unroll
  for (int h = 0; h < 4; ++h) as[h] = a_src[(size_t)row * 4 + h];
  const unsigned* arow = adj + (size_t)row * 32;
  float4* out = reinterpret_cast<float4*>(attn) + (size_t)row * N_;

#pragma unroll
  for (int it = 0; it < 4; ++it) {
    int wj = (wave << 2) + it;           // word index 0..31
    unsigned word = arow[wj];
    int j = (wj << 5) + lane;
    float4 v = make_float4(0.f, 0.f, 0.f, 0.f);
    if ((word >> lane) & 1u) {
      size_t cj = ((size_t)(c << 10) + j) * 4;
      float e0 = __expf(lrelu(as[0] + a_dst[cj + 0]) - colm[cj + 0]) * colinv[cj + 0];
      float e1 = __expf(lrelu(as[1] + a_dst[cj + 1]) - colm[cj + 1]) * colinv[cj + 1];
      float e2 = __expf(lrelu(as[2] + a_dst[cj + 2]) - colm[cj + 2]) * colinv[cj + 2];
      float e3 = __expf(lrelu(as[3] + a_dst[cj + 3]) - colm[cj + 3]) * colinv[cj + 3];
      v = make_float4(e0, e1, e2, e3);
    }
    out[(size_t)(wj << 5) + lane] = v;
  }
}

// ---------------------------------------------------------------------------
// K6: sparse aggregation. One block per dst column (c*N+j):
//   out[c,j,tf] = 0.25 * sum_{i in col} sum_h attn[c,i,j,h] * feat[c,i,h,tf]
// Deterministic compaction (prefix popcount) -> ascending-i summation order.
// ---------------------------------------------------------------------------
__global__ __launch_bounds__(256) void aggregate_kernel(const unsigned* __restrict__ adjT,
                                                        const float* __restrict__ attn,
                                                        const float* __restrict__ feat,
                                                        float* __restrict__ out) {
  __shared__ unsigned swords[32];
  __shared__ int      slist[N_];
  __shared__ int      scnt;
  int col = blockIdx.x;                  // c*N + j
  int c   = col >> 10;
  int j   = col & 1023;
  int tid = threadIdx.x;

  if (tid < 32) swords[tid] = adjT[(size_t)col * 32 + tid];
  __syncthreads();
  if (tid < 32) {
    int pre = 0;
    for (int w = 0; w < tid; ++w) pre += __popc(swords[w]);
    unsigned word = swords[tid];
    int k = pre;
    while (word) { int b = __ffs(word) - 1; word &= word - 1; slist[k++] = (tid << 5) + b; }
    if (tid == 31) scnt = k;
  }
  __syncthreads();
  int cnt = scnt;

  const float*  featc = feat + (size_t)c * N_ * (H_ * TF_);
  const float4* attnc = reinterpret_cast<const float4*>(attn)
                        + ((size_t)c * N_) * N_ + j;   // float4 index = (c*N+i)*N + j
  float acc0 = 0.f, acc1 = 0.f;
  for (int e = 0; e < cnt; ++e) {
    int i = slist[e];
    float4 w = attnc[(size_t)i * N_];
    const float* fp = featc + (size_t)i * (H_ * TF_);
    acc0 += w.x * fp[tid]        + w.y * fp[512 + tid]
          + w.z * fp[1024 + tid] + w.w * fp[1536 + tid];
    acc1 += w.x * fp[256 + tid]        + w.y * fp[512 + 256 + tid]
          + w.z * fp[1024 + 256 + tid] + w.w * fp[1536 + 256 + tid];
  }
  out[(size_t)col * TF_ + tid]       = acc0 * 0.25f;
  out[(size_t)col * TF_ + 256 + tid] = acc1 * 0.25f;
}

// ---------------------------------------------------------------------------
// Host launcher
// ---------------------------------------------------------------------------
extern "C" void kernel_launch(void* const* d_in, const int* in_sizes, int n_in,
                              void* d_out, int out_size, void* d_ws, size_t ws_size,
                              hipStream_t stream) {
  (void)n_in; (void)out_size; (void)ws_size;
  const int*   edges      = (const int*)d_in[0];
  const float* node_feats = (const float*)d_in[1];
  const float* fc         = (const float*)d_in[2];
  const float* attn_s     = (const float*)d_in[3];
  const float* attn_d     = (const float*)d_in[4];

  float* out      = (float*)d_out;
  float* attn_out = out + (size_t)C_ * N_ * T_ * F_;   // 4,194,304 floats in

  // Workspace layout (~66.5 MB)
  char* ws = (char*)d_ws;
  float*    feat   = (float*)(ws);                          // 64 MB
  float*    a_src  = (float*)(ws + 67108864);               // 128 KB
  float*    a_dst  = (float*)(ws + 67239936);               // 128 KB
  float*    colm   = (float*)(ws + 67371008);               // 128 KB
  float*    colinv = (float*)(ws + 67502080);               // 128 KB
  unsigned* adj    = (unsigned*)(ws + 67633152);            // 1 MB
  unsigned* adjT   = (unsigned*)(ws + 68681728);            // 1 MB (contiguous after adj)

  int E = in_sizes[0] / 2;

  hipMemsetAsync(adj, 0, 2u * 1048576u, stream);            // adj + adjT
  build_adj_kernel<<<(E + 255) / 256, 256, 0, stream>>>(edges, adj, adjT, E);
  gemm_feat_kernel<<<C_ * T_ * 16, 256, 0, stream>>>(node_feats, fc, feat);
  attn_vec_kernel<<<(CN_ * H_) / 8, 256, 0, stream>>>(feat, attn_s, attn_d, a_src, a_dst);
  colstats_kernel<<<CN_ / 8, 256, 0, stream>>>(adjT, a_src, a_dst, colm, colinv);
  attn_write_kernel<<<CN_, 256, 0, stream>>>(adj, a_src, a_dst, colm, colinv, attn_out);
  aggregate_kernel<<<CN_, 256, 0, stream>>>(adjT, attn_out, feat, out);
}
